// PointNet2Seg_46712064311941
// MI455X (gfx1250) — compile-verified, matrix operand order checked by reading, not run on hardware
//
#include <hip/hip_runtime.h>
#include <hip/hip_bf16.h>

typedef __attribute__((ext_vector_type(16))) _Float16 v16h;
typedef __attribute__((ext_vector_type(8)))  float    v8f;

static constexpr int CB = 8;      // batch
static constexpr int CN = 16384;  // points
static constexpr int CS1 = 1024;  // sa1 centroids
static constexpr int CS2 = 256;   // sa2 centroids
static constexpr int CK = 32;     // nsample
static constexpr int NCLS = 13;

// ---------------------------------------------------------------------------
// (B,3,N) -> (B,N,3)
__global__ void transpose_xyz_kernel(const float* __restrict__ in, float* __restrict__ out, int total /*B*N*/) {
    int t = blockIdx.x * blockDim.x + threadIdx.x;
    if (t >= total) return;
    int b = t / CN, p = t % CN;
    #pragma unroll
    for (int c = 0; c < 3; ++c)
        out[(size_t)t * 3 + c] = in[((size_t)b * 3 + c) * CN + p];
}

// ---------------------------------------------------------------------------
// Farthest point sampling: one workgroup per batch; xyz + running min-dist
// fully LDS-resident (CDNA5: 320KB/WGP). 1024-thread tree argmax per step.
__global__ void fps_kernel(const float* __restrict__ xyz /*B,n,3*/, int* __restrict__ outIdx /*B,S*/,
                           int n, int S) {
    extern __shared__ char dynsm[];
    float* sx = (float*)dynsm;
    float* sy = sx + n;
    float* sz = sy + n;
    float* sd = sz + n;
    float* rv = sd + n;                      // blockDim vals
    int*   ri = (int*)(rv + blockDim.x);     // blockDim idxs
    int b = blockIdx.x;
    const float* base = xyz + (size_t)b * n * 3;
    for (int i = threadIdx.x; i < n; i += blockDim.x) {
        sx[i] = base[i * 3 + 0];
        sy[i] = base[i * 3 + 1];
        sz[i] = base[i * 3 + 2];
        sd[i] = 1e10f;
    }
    __syncthreads();
    int far = 0;
    for (int it = 0; it < S; ++it) {
        if (threadIdx.x == 0) outIdx[b * S + it] = far;
        float cx = sx[far], cy = sy[far], cz = sz[far];
        float best = -1.0f; int bidx = 0x7fffffff;
        for (int i = threadIdx.x; i < n; i += blockDim.x) {
            float dx = sx[i] - cx, dy = sy[i] - cy, dz = sz[i] - cz;
            float d = dx * dx + dy * dy + dz * dz;
            float nd = fminf(sd[i], d);
            sd[i] = nd;
            if (nd > best) { best = nd; bidx = i; }
        }
        rv[threadIdx.x] = best; ri[threadIdx.x] = bidx;
        __syncthreads();
        for (int off = blockDim.x >> 1; off > 0; off >>= 1) {
            if (threadIdx.x < off) {
                float ov = rv[threadIdx.x + off]; int oi = ri[threadIdx.x + off];
                if (ov > rv[threadIdx.x] || (ov == rv[threadIdx.x] && oi < ri[threadIdx.x])) {
                    rv[threadIdx.x] = ov; ri[threadIdx.x] = oi;
                }
            }
            __syncthreads();
        }
        far = ri[0];
        __syncthreads();
    }
}

// ---------------------------------------------------------------------------
__global__ void gather_xyz_kernel(const float* __restrict__ xyz /*B,n,3*/, const int* __restrict__ idx /*B,S*/,
                                  float* __restrict__ out /*B,S,3*/, int n, int S) {
    int b = blockIdx.y;
    int s = blockIdx.x * blockDim.x + threadIdx.x;
    if (s >= S) return;
    int p = idx[b * S + s];
    #pragma unroll
    for (int c = 0; c < 3; ++c)
        out[((size_t)b * S + s) * 3 + c] = xyz[((size_t)b * n + p) * 3 + c];
}

// ---------------------------------------------------------------------------
// Ball query: first K in-radius indices in ascending order (== ref top-k of idx).
__global__ void ball_query_kernel(const float* __restrict__ xyz /*B,n,3*/, const float* __restrict__ cxyz /*B,S,3*/,
                                  int* __restrict__ gidx /*B,S,K*/, int n, int S, float r2) {
    const int TILE = 1024;
    __shared__ float tx[TILE], ty[TILE], tz[TILE];
    int b = blockIdx.y;
    int s = blockIdx.x * blockDim.x + threadIdx.x;
    bool active = s < S;
    float cx = 0, cy = 0, cz = 0;
    int* myIdx = gidx + ((size_t)b * S + (active ? s : 0)) * CK;
    if (active) {
        cx = cxyz[((size_t)b * S + s) * 3 + 0];
        cy = cxyz[((size_t)b * S + s) * 3 + 1];
        cz = cxyz[((size_t)b * S + s) * 3 + 2];
    }
    int cnt = 0;
    for (int t0 = 0; t0 < n; t0 += TILE) {
        int lim = min(TILE, n - t0);
        for (int i = threadIdx.x; i < lim; i += blockDim.x) {
            tx[i] = xyz[((size_t)b * n + t0 + i) * 3 + 0];
            ty[i] = xyz[((size_t)b * n + t0 + i) * 3 + 1];
            tz[i] = xyz[((size_t)b * n + t0 + i) * 3 + 2];
        }
        __syncthreads();
        if (active && cnt < CK) {
            for (int i = 0; i < lim && cnt < CK; ++i) {
                float dx = tx[i] - cx, dy = ty[i] - cy, dz = tz[i] - cz;
                if (dx * dx + dy * dy + dz * dz <= r2) myIdx[cnt++] = t0 + i;
            }
        }
        __syncthreads();
    }
    if (active) {
        int f = (cnt == 0) ? 0 : myIdx[0];
        for (int i = (cnt == 0) ? 0 : cnt; i < CK; ++i) myIdx[i] = f;
    }
}

// ---------------------------------------------------------------------------
// grouped[b,s,k,:] = concat(xyz[gidx]-centroid, feats[gidx])
__global__ void group_kernel(const float* __restrict__ xyz /*B,n,3*/, const float* __restrict__ feats /*B,n,F*/,
                             const float* __restrict__ cxyz /*B,S,3*/, const int* __restrict__ gidx,
                             float* __restrict__ out /*B,S,K,3+F*/, int n, int S, int F, int total) {
    int t = blockIdx.x * blockDim.x + threadIdx.x;
    if (t >= total) return;
    int s = (t / CK) % S;
    int b = t / (CK * S);
    int p = gidx[t];
    int stride = 3 + F;
    float* o = out + (size_t)t * stride;
    #pragma unroll
    for (int c = 0; c < 3; ++c)
        o[c] = xyz[((size_t)b * n + p) * 3 + c] - cxyz[((size_t)b * S + s) * 3 + c];
    for (int f = 0; f < F; ++f)
        o[3 + f] = feats[((size_t)b * n + p) * F + f];
}

// ---------------------------------------------------------------------------
// Fused GEMM (out = A(MxCin) * W(OxCin)^T) + bias + relu/BN/relu epilogue.
// flags: 1 = relu after bias, 2 = batchnorm, 4 = final relu.
// 8 waves/block, 128 rows/block, 32 output cols per pass (2 accumulators per
// wave -> A-fragment reused across two WMMAs). A + 32-row W panel staged
// f32->f16 in LDS; float4 global staging when Cin % 4 == 0.
__global__ void gemm_wmma_kernel(const float* __restrict__ A, const float* __restrict__ W,
                                 const float* __restrict__ bias, const float* __restrict__ gamma,
                                 const float* __restrict__ beta, const float* __restrict__ mean,
                                 const float* __restrict__ var, float* __restrict__ out,
                                 int M, int Cin, int CinP, int O, int flags) {
    extern __shared__ char dynsm[];
    _Float16* As = (_Float16*)dynsm;           // 128 x CinP
    _Float16* Ws = As + 128 * CinP;            // 32 x CinP
    int m0 = blockIdx.x * 128;
    int tid = threadIdx.x;
    int wave = tid >> 5;
    int lane = tid & 31;
    int mloc = wave * 16 + (lane & 15);
    int nlane = lane & 15;
    int khalf = lane >> 4;
    bool vec4 = (Cin == CinP) && ((Cin & 3) == 0);

    // ---- stage A panel ----
    if (vec4) {
        int q = Cin >> 2;
        for (int idx = tid; idx < 128 * q; idx += 256) {
            int r = idx / q, c4 = idx % q;
            int gm = m0 + r;
            float4 v = (gm < M) ? ((const float4*)A)[(size_t)gm * q + c4]
                                : make_float4(0.f, 0.f, 0.f, 0.f);
            _Float16* dst = As + (size_t)r * CinP + 4 * c4;
            dst[0] = (_Float16)v.x; dst[1] = (_Float16)v.y;
            dst[2] = (_Float16)v.z; dst[3] = (_Float16)v.w;
        }
    } else {
        for (int idx = tid; idx < 128 * CinP; idx += 256) {
            int r = idx / CinP, c = idx % CinP;
            int gm = m0 + r;
            float v = (gm < M && c < Cin) ? A[(size_t)gm * Cin + c] : 0.0f;
            As[idx] = (_Float16)v;
        }
    }
    __syncthreads();

    for (int o0 = 0; o0 < O; o0 += 32) {
        // ---- stage 32-row W panel (zero-padded past O / Cin) ----
        if (vec4) {
            int q = Cin >> 2;
            for (int idx = tid; idx < 32 * q; idx += 256) {
                int r = idx / q, c4 = idx % q;
                int go = o0 + r;
                float4 v = (go < O) ? ((const float4*)W)[(size_t)go * q + c4]
                                    : make_float4(0.f, 0.f, 0.f, 0.f);
                _Float16* dst = Ws + (size_t)r * CinP + 4 * c4;
                dst[0] = (_Float16)v.x; dst[1] = (_Float16)v.y;
                dst[2] = (_Float16)v.z; dst[3] = (_Float16)v.w;
            }
        } else {
            for (int idx = tid; idx < 32 * CinP; idx += 256) {
                int r = idx / CinP, c = idx % CinP;
                int go = o0 + r;
                float v = (go < O && c < Cin) ? W[(size_t)go * Cin + c] : 0.0f;
                Ws[idx] = (_Float16)v;
            }
        }
        __syncthreads();

        v8f acc0 = {};
        v8f acc1 = {};
        for (int k0 = 0; k0 < CinP; k0 += 32) {
            v16h af, bf0, bf1;
            #pragma unroll
            for (int j = 0; j < 8; ++j) {
                int kb = k0 + (j >> 2) * 16 + khalf * 8 + (j & 3) * 2;
                af[2 * j]      = As[mloc * CinP + kb];
                af[2 * j + 1]  = As[mloc * CinP + kb + 1];
                bf0[2 * j]     = Ws[nlane * CinP + kb];
                bf0[2 * j + 1] = Ws[nlane * CinP + kb + 1];
                bf1[2 * j]     = Ws[(16 + nlane) * CinP + kb];
                bf1[2 * j + 1] = Ws[(16 + nlane) * CinP + kb + 1];
            }
            acc0 = __builtin_amdgcn_wmma_f32_16x16x32_f16(false, af, false, bf0, (short)0, acc0, false, false);
            acc1 = __builtin_amdgcn_wmma_f32_16x16x32_f16(false, af, false, bf1, (short)0, acc1, false, false);
        }

        // ---- fused epilogue for both 16-col tiles ----
        #pragma unroll
        for (int half = 0; half < 2; ++half) {
            int nglob = o0 + half * 16 + nlane;
            if (nglob < O) {
                float bb = bias[nglob];
                float g = 0, bt = 0, mn = 0, iv = 0;
                if (flags & 2) {
                    g = gamma[nglob]; bt = beta[nglob]; mn = mean[nglob];
                    iv = rsqrtf(var[nglob] + 1e-5f);
                }
                #pragma unroll
                for (int r = 0; r < 8; ++r) {
                    int gm = m0 + wave * 16 + r + 8 * khalf;
                    if (gm < M) {
                        float y = (half ? acc1[r] : acc0[r]) + bb;
                        if (flags & 1) y = fmaxf(y, 0.0f);
                        if (flags & 2) y = (y - mn) * iv * g + bt;
                        if (flags & 4) y = fmaxf(y, 0.0f);
                        out[(size_t)gm * O + nglob] = y;
                    }
                }
            }
        }
        __syncthreads();
    }
}

// ---------------------------------------------------------------------------
__global__ void maxpool_kernel(const float* __restrict__ in /*B,S,K,O*/, float* __restrict__ out /*B,S,O*/,
                               int O, int total /*B*S*O*/) {
    int t = blockIdx.x * blockDim.x + threadIdx.x;
    if (t >= total) return;
    int o = t % O;
    int bs = t / O;
    float m = -1e30f;
    for (int k = 0; k < CK; ++k)
        m = fmaxf(m, in[((size_t)bs * CK + k) * O + o]);
    out[t] = m;
}

// ---------------------------------------------------------------------------
// 3-NN (candidates staged in LDS) + inverse-distance weights.
__global__ void three_nn_kernel(const float* __restrict__ xyz1 /*B,M1,3*/, const float* __restrict__ xyz2 /*B,M2,3*/,
                                int* __restrict__ nnIdx /*B,M1,3*/, float* __restrict__ nnW /*B,M1,3*/,
                                int M1, int M2) {
    __shared__ float sx[1024], sy[1024], sz[1024];
    int b = blockIdx.y;
    int nq = blockIdx.x * blockDim.x + threadIdx.x;
    for (int i = threadIdx.x; i < M2; i += blockDim.x) {
        sx[i] = xyz2[((size_t)b * M2 + i) * 3 + 0];
        sy[i] = xyz2[((size_t)b * M2 + i) * 3 + 1];
        sz[i] = xyz2[((size_t)b * M2 + i) * 3 + 2];
    }
    __syncthreads();
    if (nq >= M1) return;
    float px = xyz1[((size_t)b * M1 + nq) * 3 + 0];
    float py = xyz1[((size_t)b * M1 + nq) * 3 + 1];
    float pz = xyz1[((size_t)b * M1 + nq) * 3 + 2];
    float d0 = 1e30f, d1 = 1e30f, d2 = 1e30f;
    int i0 = 0, i1 = 0, i2 = 0;
    for (int j = 0; j < M2; ++j) {
        float dx = sx[j] - px, dy = sy[j] - py, dz = sz[j] - pz;
        float d = dx * dx + dy * dy + dz * dz;
        if (d < d0)      { d2 = d1; i2 = i1; d1 = d0; i1 = i0; d0 = d; i0 = j; }
        else if (d < d1) { d2 = d1; i2 = i1; d1 = d; i1 = j; }
        else if (d < d2) { d2 = d; i2 = j; }
    }
    float r0 = 1.0f / (d0 + 1e-8f), r1 = 1.0f / (d1 + 1e-8f), r2 = 1.0f / (d2 + 1e-8f);
    float inv = 1.0f / (r0 + r1 + r2);
    size_t o = ((size_t)b * M1 + nq) * 3;
    nnIdx[o + 0] = i0; nnIdx[o + 1] = i1; nnIdx[o + 2] = i2;
    nnW[o + 0] = r0 * inv; nnW[o + 1] = r1 * inv; nnW[o + 2] = r2 * inv;
}

// ---------------------------------------------------------------------------
__global__ void interp_kernel(const float* __restrict__ pts2 /*B,M2,C*/, const int* __restrict__ nnIdx,
                              const float* __restrict__ nnW, float* __restrict__ out /*B,M1,Ctot*/,
                              int M1, int M2, int C, int Ctot) {
    int bn = blockIdx.x;
    int b = bn / M1;
    size_t o3 = (size_t)bn * 3;
    int i0 = nnIdx[o3], i1 = nnIdx[o3 + 1], i2 = nnIdx[o3 + 2];
    float w0 = nnW[o3], w1 = nnW[o3 + 1], w2 = nnW[o3 + 2];
    const float* p0 = pts2 + ((size_t)b * M2 + i0) * C;
    const float* p1 = pts2 + ((size_t)b * M2 + i1) * C;
    const float* p2 = pts2 + ((size_t)b * M2 + i2) * C;
    for (int c = threadIdx.x; c < C; c += blockDim.x)
        out[(size_t)bn * Ctot + c] = w0 * p0[c] + w1 * p1[c] + w2 * p2[c];
}

// concat helper: write pts1 features into columns [coff, coff+C1)
__global__ void copy_feats_kernel(const float* __restrict__ pts1 /*BN,C1*/, float* __restrict__ out /*BN,Ctot*/,
                                  int C1, int Ctot, int coff) {
    int bn = blockIdx.x;
    for (int c = threadIdx.x; c < C1; c += blockDim.x)
        out[(size_t)bn * Ctot + coff + c] = pts1[(size_t)bn * C1 + c];
}

// ---------------------------------------------------------------------------
extern "C" void kernel_launch(void* const* d_in, const int* in_sizes, int n_in,
                              void* d_out, int out_size, void* d_ws, size_t ws_size,
                              hipStream_t stream) {
    (void)in_sizes; (void)n_in; (void)out_size; (void)ws_size;
    auto F = [&](int i) { return (const float*)d_in[i]; };
    const float* xyz_raw = F(0);  // (B,3,N)

    // ---- workspace carve ----
    char* ws = (char*)d_ws;
    size_t off = 0;
    auto take = [&](size_t bytes) -> void* {
        void* p = ws + off;
        off += (bytes + 255) & ~(size_t)255;
        return p;
    };
    float* xyzT   = (float*)take((size_t)CB * CN * 3 * 4);
    int*   fps1   = (int*)  take((size_t)CB * CS1 * 4);
    float* l1xyz  = (float*)take((size_t)CB * CS1 * 3 * 4);
    int*   gidx1  = (int*)  take((size_t)CB * CS1 * CK * 4);
    int*   fps2   = (int*)  take((size_t)CB * CS2 * 4);
    float* l2xyz  = (float*)take((size_t)CB * CS2 * 3 * 4);
    int*   gidx2  = (int*)  take((size_t)CB * CS2 * CK * 4);
    float* l1feat = (float*)take((size_t)CB * CS1 * 128 * 4);
    float* l2feat = (float*)take((size_t)CB * CS2 * 256 * 4);
    int*   nnI1   = (int*)  take((size_t)CB * CS1 * 3 * 4);
    float* nnW1   = (float*)take((size_t)CB * CS1 * 3 * 4);
    float* l1new  = (float*)take((size_t)CB * CS1 * 256 * 4);
    int*   nnI2   = (int*)  take((size_t)CB * CN * 3 * 4);
    float* nnW2   = (float*)take((size_t)CB * CN * 3 * 4);
    float* buf0   = (float*)take((size_t)CB * CN * 256 * 4);
    float* buf1   = (float*)take((size_t)CB * CN * 256 * 4);

    auto gemm = [&](const float* Ain, int wIdx, float* outp, int M, int Cin, int O, int flags) {
        int CinP = (Cin + 31) & ~31;
        size_t shmem = (size_t)(128 + 32) * CinP * 2;
        const float* gp = nullptr, *btp = nullptr, *mp = nullptr, *vp = nullptr;
        if (flags & 2) { gp = F(wIdx + 2); btp = F(wIdx + 3); mp = F(wIdx + 4); vp = F(wIdx + 5); }
        gemm_wmma_kernel<<<dim3((M + 127) / 128), dim3(256), shmem, stream>>>(
            Ain, F(wIdx), F(wIdx + 1), gp, btp, mp, vp, outp, M, Cin, CinP, O, flags);
    };

    // ---- input transpose: (B,3,N) -> (B,N,3) ----
    transpose_xyz_kernel<<<(CB * CN + 255) / 256, 256, 0, stream>>>(xyz_raw, xyzT, CB * CN);

    // =================== SA1 (npoint=1024, r=0.1, K=32) ===================
    {
        size_t sh = (size_t)(4 * CN + 1024) * 4 + 1024 * 4;
        fps_kernel<<<CB, 1024, sh, stream>>>(xyzT, fps1, CN, CS1);
        gather_xyz_kernel<<<dim3(CS1 / 256, CB), 256, 0, stream>>>(xyzT, fps1, l1xyz, CN, CS1);
        ball_query_kernel<<<dim3(CS1 / 256, CB), 256, 0, stream>>>(xyzT, l1xyz, gidx1, CN, CS1, 0.01f);
        int rows = CB * CS1 * CK;
        group_kernel<<<(rows + 255) / 256, 256, 0, stream>>>(xyzT, nullptr, l1xyz, gidx1, buf0, CN, CS1, 0, rows);
        gemm(buf0, 1,  buf1, rows, 3,   64,  7);   // relu -> bn -> relu
        gemm(buf1, 7,  buf0, rows, 64,  64,  7);
        gemm(buf0, 13, buf1, rows, 64,  128, 7);
        int tot = CB * CS1 * 128;
        maxpool_kernel<<<(tot + 255) / 256, 256, 0, stream>>>(buf1, l1feat, 128, tot);
    }

    // =================== SA2 (npoint=256, r=0.2, K=32) ===================
    {
        size_t sh = (size_t)(4 * CS1 + 1024) * 4 + 1024 * 4;
        fps_kernel<<<CB, 1024, sh, stream>>>(l1xyz, fps2, CS1, CS2);
        gather_xyz_kernel<<<dim3(1, CB), 256, 0, stream>>>(l1xyz, fps2, l2xyz, CS1, CS2);
        ball_query_kernel<<<dim3(1, CB), 256, 0, stream>>>(l1xyz, l2xyz, gidx2, CS1, CS2, 0.04f);
        int rows = CB * CS2 * CK;
        group_kernel<<<(rows + 255) / 256, 256, 0, stream>>>(l1xyz, l1feat, l2xyz, gidx2, buf0, CS1, CS2, 128, rows);
        gemm(buf0, 19, buf1, rows, 131, 128, 7);
        gemm(buf1, 25, buf0, rows, 128, 128, 7);
        gemm(buf0, 31, buf1, rows, 128, 256, 7);
        int tot = CB * CS2 * 256;
        maxpool_kernel<<<(tot + 255) / 256, 256, 0, stream>>>(buf1, l2feat, 256, tot);
    }

    // =================== FP1: (l1_xyz <- l2) + concat l1 ===================
    {
        three_nn_kernel<<<dim3((CS1 + 255) / 256, CB), 256, 0, stream>>>(l1xyz, l2xyz, nnI1, nnW1, CS1, CS2);
        interp_kernel<<<CB * CS1, 256, 0, stream>>>(l2feat, nnI1, nnW1, buf0, CS1, CS2, 256, 384);
        copy_feats_kernel<<<CB * CS1, 128, 0, stream>>>(l1feat, buf0, 128, 384, 256);
        int rows = CB * CS1;
        gemm(buf0, 37, buf1, rows, 384, 256, 6);   // bn -> relu
        gemm(buf1, 43, l1new, rows, 256, 256, 4);  // relu only
    }

    // =================== FP2: (xyz <- l1new) ===================
    {
        three_nn_kernel<<<dim3((CN + 255) / 256, CB), 256, 0, stream>>>(xyzT, l1xyz, nnI2, nnW2, CN, CS1);
        interp_kernel<<<CB * CN, 256, 0, stream>>>(l1new, nnI2, nnW2, buf0, CN, CS1, 256, 256);
        int rows = CB * CN;
        gemm(buf0, 45, buf1, rows, 256, 256, 6);
        gemm(buf1, 51, buf0, rows, 256, 128, 4);
    }

    // =================== Head ===================
    {
        int rows = CB * CN;
        gemm(buf0, 53, buf1, rows, 128, 128, 6);
        gemm(buf1, 59, (float*)d_out, rows, 128, NCLS, 0);  // linear, no relu
    }
}
